// RelPositionMultiHeadedAttention_10488310137460
// MI455X (gfx1250) — matrix operand order repaired and checked by reading, not compiled, not run on hardware
//
#include <hip/hip_runtime.h>
#include <hip/hip_bf16.h>
#include <math.h>

// ---------------------------------------------------------------------------
// RelPositionMultiHeadedAttention forward for MI455X (gfx1250, wave32, WMMA)
// B=4, T=2048, D=512, H=8, DK=64.  All GEMMs via v_wmma_f32_16x16x32_bf16.
// Attention uses shift-invariant softmax (fixed exponent offset) and computes
// row sums with a ones-matrix WMMA -> no cross-lane shuffles at all.
// ---------------------------------------------------------------------------

typedef __bf16 v16bf __attribute__((ext_vector_type(16)));
typedef float  v8f   __attribute__((ext_vector_type(8)));

#define DEV __device__ __forceinline__

static constexpr int Bn = 4, Tn = 2048, Dn = 512, Hn = 8, DKn = 64;

DEV v8f wmma_bf16(v16bf a, v16bf b, v8f c) {
  // (neg_a, A, neg_b, B, c_mod, C, reuse_a, reuse_b)
  return __builtin_amdgcn_wmma_f32_16x16x32_bf16(false, a, false, b, (short)0, c,
                                                 false, false);
}

// --- A fragment (16x32 MxK, bf16) -----------------------------------------
// lanes 0-15: M=lane, elems 0..15 -> K=k0..k0+15
// lanes 16-31: M=lane-16, elems 0..15 -> K=k0+16..k0+31
DEV v16bf a_frag_f32(const float* __restrict__ X, int ld, int m0, int k0) {
  const int lane = threadIdx.x & 31;
  const int hf = lane >> 4;
  const float* p = X + (size_t)(m0 + (lane & 15)) * ld + k0 + hf * 16;
  v16bf a;
#pragma unroll
  for (int i = 0; i < 8; ++i) a[i] = (__bf16)p[i];
#pragma unroll
  for (int i = 0; i < 8; ++i) a[8 + i] = (__bf16)p[8 + i];
  return a;
}

DEV v16bf a_frag_bf16(const __bf16* __restrict__ X, int ld, int m0, int k0) {
  const int lane = threadIdx.x & 31;
  const int hf = lane >> 4;
  const __bf16* p = X + (size_t)(m0 + (lane & 15)) * ld + k0 + hf * 16;
  v16bf a;
#pragma unroll
  for (int i = 0; i < 8; ++i) a[i] = p[i];
#pragma unroll
  for (int i = 0; i < 8; ++i) a[8 + i] = p[8 + i];
  return a;
}

// --- B fragment (32x16 KxN, bf16); B[k][n] read from row-major M[n][k] -----
// lane n = n0 + lane&15; elems j -> K = k0 + j + 16*(lane>=16): contiguous.
DEV v16bf b_frag_f32(const float* __restrict__ W, int ld, int n0, int k0) {
  const int lane = threadIdx.x & 31;
  const int hf = lane >> 4;
  const float* p = W + (size_t)(n0 + (lane & 15)) * ld + k0 + hf * 16;
  v16bf b;
#pragma unroll
  for (int i = 0; i < 16; ++i) b[i] = (__bf16)p[i];
  return b;
}

DEV v16bf b_frag_bf16(const __bf16* __restrict__ W, int ld, int n0, int k0) {
  const int lane = threadIdx.x & 31;
  const int hf = lane >> 4;
  const __bf16* p = W + (size_t)(n0 + (lane & 15)) * ld + k0 + hf * 16;
  v16bf b;
#pragma unroll
  for (int i = 0; i < 16; ++i) b[i] = p[i];
  return b;
}

// --- 16x64 wave-tile GEMM: Y = X(MxK) * W^T (W is NxK row-major) -----------
template <typename T>
DEV void gemm_16x64(const T* __restrict__ X, const float* __restrict__ W,
                    int K, int m0, int n0, v8f acc[4]) {
  for (int k0 = 0; k0 < K; k0 += 32) {
    v16bf a;
    if constexpr (sizeof(T) == 4)
      a = a_frag_f32((const float*)X, K, m0, k0);
    else
      a = a_frag_bf16((const __bf16*)X, K, m0, k0);
#pragma unroll
    for (int j = 0; j < 4; ++j) {
      v16bf b = b_frag_f32(W, K, n0 + j * 16, k0);
      acc[j] = wmma_bf16(a, b, acc[j]);
    }
  }
}

// ---------------------------------------------------------------------------
// Projection kernels. 8 waves/block, each wave = one 16x64 output tile.
// C/D layout: lane<16 -> row m0+r, lane>=16 -> row m0+r+8; col = n0+(lane&15).
// ---------------------------------------------------------------------------
__global__ __launch_bounds__(256) void proj_q_kernel(
    const float* __restrict__ query, const float* __restrict__ Wq,
    const float* __restrict__ bq, const float* __restrict__ ubias,
    const float* __restrict__ vbias, __bf16* __restrict__ Qu,
    __bf16* __restrict__ Qv) {
  const int wave = blockIdx.x * 8 + (threadIdx.x >> 5);
  const int m0 = (wave >> 3) * 16;
  const int n0 = (wave & 7) * 64;
  v8f acc[4] = {};
  gemm_16x64(query, Wq, Dn, m0, n0, acc);
  const int lane = threadIdx.x & 31, hf = lane >> 4;
#pragma unroll
  for (int j = 0; j < 4; ++j) {
    const int n = n0 + j * 16 + (lane & 15);
    const int h = n >> 6, dk = n & 63;
    const float bb = bq[n], uu = ubias[n], vv = vbias[n];
#pragma unroll
    for (int r = 0; r < 8; ++r) {
      const int m = m0 + r + hf * 8;
      const int b = m >> 11, t = m & (Tn - 1);
      const size_t idx = (((size_t)(b * Hn + h) * Tn) + t) * DKn + dk;
      const float y = acc[j][r] + bb;
      Qu[idx] = (__bf16)(y + uu);
      Qv[idx] = (__bf16)(y + vv);
    }
  }
}

__global__ __launch_bounds__(256) void proj_k_kernel(
    const float* __restrict__ key, const float* __restrict__ Wk,
    const float* __restrict__ bk, __bf16* __restrict__ Kh) {
  const int wave = blockIdx.x * 8 + (threadIdx.x >> 5);
  const int m0 = (wave >> 3) * 16;
  const int n0 = (wave & 7) * 64;
  v8f acc[4] = {};
  gemm_16x64(key, Wk, Dn, m0, n0, acc);
  const int lane = threadIdx.x & 31, hf = lane >> 4;
#pragma unroll
  for (int j = 0; j < 4; ++j) {
    const int n = n0 + j * 16 + (lane & 15);
    const int h = n >> 6, dk = n & 63;
    const float bb = bk[n];
#pragma unroll
    for (int r = 0; r < 8; ++r) {
      const int m = m0 + r + hf * 8;
      const int b = m >> 11, t = m & (Tn - 1);
      Kh[(((size_t)(b * Hn + h) * Tn) + t) * DKn + dk] = (__bf16)(acc[j][r] + bb);
    }
  }
}

// V stored transposed (b,h,dk,s) so attention's B-fragments are contiguous.
__global__ __launch_bounds__(256) void proj_v_kernel(
    const float* __restrict__ value, const float* __restrict__ Wv,
    const float* __restrict__ bv, __bf16* __restrict__ Vt) {
  const int wave = blockIdx.x * 8 + (threadIdx.x >> 5);
  const int m0 = (wave >> 3) * 16;
  const int n0 = (wave & 7) * 64;
  v8f acc[4] = {};
  gemm_16x64(value, Wv, Dn, m0, n0, acc);
  const int lane = threadIdx.x & 31, hf = lane >> 4;
#pragma unroll
  for (int j = 0; j < 4; ++j) {
    const int n = n0 + j * 16 + (lane & 15);
    const int h = n >> 6, dk = n & 63;
    const float bb = bv[n];
#pragma unroll
    for (int r = 0; r < 8; ++r) {
      const int m = m0 + r + hf * 8;
      const int b = m >> 11, t = m & (Tn - 1);
      Vt[(((size_t)(b * Hn + h) * DKn) + dk) * Tn + t] = (__bf16)(acc[j][r] + bb);
    }
  }
}

__global__ __launch_bounds__(256) void proj_p_kernel(
    const float* __restrict__ pos, const float* __restrict__ Wp,
    __bf16* __restrict__ Pp) {
  const int wave = blockIdx.x * 8 + (threadIdx.x >> 5);
  const int m0 = (wave >> 3) * 16;  // m = s (pos index)
  const int n0 = (wave & 7) * 64;
  v8f acc[4] = {};
  gemm_16x64(pos, Wp, Dn, m0, n0, acc);
  const int lane = threadIdx.x & 31, hf = lane >> 4;
#pragma unroll
  for (int j = 0; j < 4; ++j) {
    const int n = n0 + j * 16 + (lane & 15);
    const int h = n >> 6, dk = n & 63;
#pragma unroll
    for (int r = 0; r < 8; ++r) {
      const int s = m0 + r + hf * 8;
      Pp[((size_t)h * Tn + s) * DKn + dk] = (__bf16)acc[j][r];
    }
  }
}

// ---------------------------------------------------------------------------
// Flash attention, one wave per (b, h, 32 query rows) = two 16-row A-tiles
// sharing all K/P/V B-fragments (26 WMMA per 12 B-fragment loads).
// softmax is shift-invariant: use fixed offset (scores are O(1) here, fp32
// exp safe to ~88), p = exp2(S*log2e/8 - 10*log2e). Row sums via ones-matrix
// WMMA (D layout broadcasts the sum to every lane of the row): no shuffles.
// ---------------------------------------------------------------------------
__global__ __launch_bounds__(256) void attn_kernel(
    const __bf16* __restrict__ Qu, const __bf16* __restrict__ Qv,
    const __bf16* __restrict__ Kh, const __bf16* __restrict__ Vt,
    const __bf16* __restrict__ Pp, const int* __restrict__ mask,
    __bf16* __restrict__ Xb) {
  __shared__ __bf16 plds[8][32 * 32];  // per-wave 32x32 P staging (2 KB each)
  const int wv = threadIdx.x >> 5;
  const int wave = blockIdx.x * 8 + wv;
  const int b = wave >> 9;          // 512 waves per batch
  const int h = (wave >> 6) & 7;    // 64 waves per head
  const int t0 = (wave & 63) * 32;  // 32 query rows per wave
  const int lane = threadIdx.x & 31, hf = lane >> 4;

  const size_t bh = (size_t)(b * Hn + h);
  const __bf16* qu = Qu + bh * Tn * DKn;
  const __bf16* qv = Qv + bh * Tn * DKn;
  const __bf16* kk = Kh + bh * Tn * DKn;
  const __bf16* vt = Vt + bh * DKn * Tn;
  const __bf16* pp = Pp + (size_t)h * Tn * DKn;
  const int* mb = mask + b * Tn;

  // Loop-invariant query fragments for the two 16-row tiles.
  v16bf au0[2], au1[2], av0[2], av1[2];
#pragma unroll
  for (int q = 0; q < 2; ++q) {
    au0[q] = a_frag_bf16(qu, DKn, t0 + q * 16, 0);
    au1[q] = a_frag_bf16(qu, DKn, t0 + q * 16, 32);
    av0[q] = a_frag_bf16(qv, DKn, t0 + q * 16, 0);
    av1[q] = a_frag_bf16(qv, DKn, t0 + q * 16, 32);
  }

  v16bf ones;
#pragma unroll
  for (int i = 0; i < 16; ++i) ones[i] = (__bf16)1.0f;

  v8f o[2][4] = {};
  v8f rsum[2] = {};

  // exp2(S/8*log2e - 10*log2e): fold scale+offset+log2e into one fma.
  constexpr float kSc = 0.125f * 1.44269504088896340736f;
  constexpr float kOff = -10.0f * 1.44269504088896340736f;

  __bf16* pb = &plds[wv][0];

  for (int s0 = 0; s0 < Tn; s0 += 32) {
    if (s0 + 32 < Tn) {  // uniform branch; prefetch next K/V tiles
      __builtin_prefetch(kk + (size_t)(s0 + 32) * DKn + (lane << 2), 0, 0);
      __builtin_prefetch(vt + (size_t)((lane & 15) << 1) * Tn + s0 + 32, 0, 0);
    }
    // scores + exp -> P staged in LDS (rows: q*16 + r + 8*hf, cols: 32 of s)
#pragma unroll
    for (int sub = 0; sub < 2; ++sub) {
      const int sc = s0 + sub * 16;
      const v16bf bk0 = b_frag_bf16(kk, DKn, sc, 0);
      const v16bf bk1 = b_frag_bf16(kk, DKn, sc, 32);
      const v16bf bp0 = b_frag_bf16(pp, DKn, sc, 0);
      const v16bf bp1 = b_frag_bf16(pp, DKn, sc, 32);
      const bool msk = (mb[sc + (lane & 15)] == 0);
#pragma unroll
      for (int q = 0; q < 2; ++q) {
        v8f s = {};
        s = wmma_bf16(au0[q], bk0, s);
        s = wmma_bf16(au1[q], bk1, s);
        s = wmma_bf16(av0[q], bp0, s);
        s = wmma_bf16(av1[q], bp1, s);
        const int row0 = q * 16 + hf * 8;
        const int col = (lane & 15) + sub * 16;
#pragma unroll
        for (int r = 0; r < 8; ++r) {
          const float p =
              msk ? 0.f : __builtin_amdgcn_exp2f(fmaf(s[r], kSc, kOff));
          pb[(row0 + r) * 32 + col] = (__bf16)p;
        }
      }
    }
    // P-fragments back from LDS; row sums via ones-WMMA; O += P . V
    const v16bf ap0 = a_frag_bf16(pb, 32, 0, 0);
    const v16bf ap1 = a_frag_bf16(pb, 32, 16, 0);
    rsum[0] = wmma_bf16(ap0, ones, rsum[0]);
    rsum[1] = wmma_bf16(ap1, ones, rsum[1]);
#pragma unroll
    for (int j = 0; j < 4; ++j) {
      const v16bf bvf = b_frag_bf16(vt, Tn, j * 16, s0);
      o[0][j] = wmma_bf16(ap0, bvf, o[0][j]);
      o[1][j] = wmma_bf16(ap1, bvf, o[1][j]);
    }
  }
  // --- normalize and store x (bf16); rsum D-layout matches o's layout ---
#pragma unroll
  for (int q = 0; q < 2; ++q) {
#pragma unroll
    for (int r = 0; r < 8; ++r) {
      const float inv = 1.f / fmaxf(rsum[q][r], 1e-30f);
      const int t = t0 + q * 16 + r + hf * 8;
#pragma unroll
      for (int j = 0; j < 4; ++j) {
        const int dk = j * 16 + (lane & 15);
        Xb[((size_t)b * Tn + t) * Dn + h * DKn + dk] = (__bf16)(o[q][j][r] * inv);
      }
    }
  }
}

// --- output projection: out = X . Wo^T + bo (fp32 out) ---------------------
__global__ __launch_bounds__(256) void proj_out_kernel(
    const __bf16* __restrict__ Xb, const float* __restrict__ Wo,
    const float* __restrict__ bo, float* __restrict__ out) {
  const int wave = blockIdx.x * 8 + (threadIdx.x >> 5);
  const int m0 = (wave >> 3) * 16;
  const int n0 = (wave & 7) * 64;
  v8f acc[4] = {};
  gemm_16x64(Xb, Wo, Dn, m0, n0, acc);
  const int lane = threadIdx.x & 31, hf = lane >> 4;
#pragma unroll
  for (int j = 0; j < 4; ++j) {
    const int n = n0 + j * 16 + (lane & 15);
    const float bb = bo[n];
#pragma unroll
    for (int r = 0; r < 8; ++r) {
      const int m = m0 + r + hf * 8;
      out[(size_t)m * Dn + n] = acc[j][r] + bb;
    }
  }
}

// ---------------------------------------------------------------------------
extern "C" void kernel_launch(void* const* d_in, const int* in_sizes, int n_in,
                              void* d_out, int out_size, void* d_ws,
                              size_t ws_size, hipStream_t stream) {
  const float* query = (const float*)d_in[0];
  const float* key   = (const float*)d_in[1];
  const float* value = (const float*)d_in[2];
  const int*   mask  = (const int*)d_in[3];
  const float* pos   = (const float*)d_in[4];
  const float* Wq = (const float*)d_in[5];
  const float* bq = (const float*)d_in[6];
  const float* Wk = (const float*)d_in[7];
  const float* bk = (const float*)d_in[8];
  const float* Wv = (const float*)d_in[9];
  const float* bv = (const float*)d_in[10];
  const float* Wo = (const float*)d_in[11];
  const float* bo = (const float*)d_in[12];
  const float* Wp = (const float*)d_in[13];
  const float* pu = (const float*)d_in[14];
  const float* pv = (const float*)d_in[15];
  float* out = (float*)d_out;

  // workspace carve (bf16): Qu,Qv,K,Vt (8MB each) + P (2MB) + X (8MB) = 42MB
  char* ws = (char*)d_ws;
  const size_t NQ = (size_t)Bn * Hn * Tn * DKn;  // 4,194,304 elems
  __bf16* Qu = (__bf16*)ws; ws += NQ * 2;
  __bf16* Qv = (__bf16*)ws; ws += NQ * 2;
  __bf16* Kh = (__bf16*)ws; ws += NQ * 2;
  __bf16* Vt = (__bf16*)ws; ws += NQ * 2;
  __bf16* Pp = (__bf16*)ws; ws += (size_t)Hn * Tn * DKn * 2;
  __bf16* Xb = (__bf16*)ws;

  dim3 blk(256);  // 8 wave32s per block
  proj_q_kernel<<<512, blk, 0, stream>>>(query, Wq, bq, pu, pv, Qu, Qv);
  proj_k_kernel<<<512, blk, 0, stream>>>(key, Wk, bk, Kh);
  proj_v_kernel<<<512, blk, 0, stream>>>(value, Wv, bv, Vt);
  proj_p_kernel<<<128, blk, 0, stream>>>(pos, Wp, Pp);
  attn_kernel<<<256, blk, 0, stream>>>(Qu, Qv, Kh, Vt, Pp, mask, Xb);
  proj_out_kernel<<<512, blk, 0, stream>>>(Xb, Wo, bo, out);
}
